// cherry_module_8048768712805
// MI455X (gfx1250) — compile-verified, matrix operand order checked by reading, not run on hardware
//
#include <hip/hip_runtime.h>
#include <hip/hip_bf16.h>
#include <stdint.h>

typedef __attribute__((ext_vector_type(16))) __bf16 v16bf;
typedef __attribute__((ext_vector_type(8)))  float  v8f;

#define N_NODES 100000
#define N_EDGES 800000
#define N_LABEL 400000
#define CH 256
#define HID1 128
#define HID2 32

__device__ __forceinline__ unsigned short f2bf(float f) {
    unsigned int u = __float_as_uint(f);
    u += 0x7FFFu + ((u >> 16) & 1u);   // round-to-nearest-even to bf16
    return (unsigned short)(u >> 16);
}

union FragAB { v16bf v; uint4 q[2]; };
union FragC  { v8f  v; float f[8]; };

// ---------------------------------------------------------------------------
// Pack a row-major fp32 weight [fin x fout] into bf16 WMMA B-fragment layout.
// Fragment (kt,nt) covers K=kt*32..+31, N=nt*16..+15; stored as
// [32 lanes][16 bf16] contiguous (512 ushorts / fragment).
// Lane layout mirrors the 16-bit A/B layout from ISA 7.12.2:
//   lanes 0-15 : K = {0..7, 16..23}; lanes 16-31 : K = {8..15, 24..31}
// ---------------------------------------------------------------------------
__global__ void pack_weight(const float* __restrict__ W,
                            unsigned short* __restrict__ out,
                            int fin, int fout) {
    int lane = threadIdx.x;                 // 32 threads / block = 1 fragment
    int ntiles = fout >> 4;
    int kt = blockIdx.x / ntiles;
    int nt = blockIdx.x % ntiles;
    int n = nt * 16 + (lane & 15);
    int hi = (lane >= 16) ? 8 : 0;
    unsigned short* dst = out + (size_t)blockIdx.x * 512 + lane * 16;
#pragma unroll
    for (int i = 0; i < 16; ++i) {
        int base = ((i < 8) ? 0 : 16) + hi;
        int off  = (i < 8) ? i : (i - 8);
        int k = kt * 32 + base + off;
        dst[i] = f2bf(W[(size_t)k * fout + n]);
    }
}

// ---------------------------------------------------------------------------
// Degree / normalization
// ---------------------------------------------------------------------------
__global__ void deg_init(float* __restrict__ deg) {
    int i = blockIdx.x * blockDim.x + threadIdx.x;
    if (i < N_NODES) deg[i] = 1.0f;                 // self loop
}
__global__ void deg_count(const int* __restrict__ ei, float* __restrict__ deg) {
    int e = blockIdx.x * blockDim.x + threadIdx.x;
    if (e < N_EDGES) atomicAdd(&deg[ei[N_EDGES + e]], 1.0f);   // dst row
}
__global__ void dinv_kernel(const float* __restrict__ deg, float* __restrict__ dinv) {
    int i = blockIdx.x * blockDim.x + threadIdx.x;
    if (i < N_NODES) dinv[i] = rsqrtf(deg[i]);
}

// ---------------------------------------------------------------------------
// h0 = x @ W_gcn  (bf16 WMMA, fp32 accumulate). One wave per 16-row tile.
// ---------------------------------------------------------------------------
__global__ __launch_bounds__(128) void gemm_h0(const float* __restrict__ x,
                                               const unsigned short* __restrict__ pW,
                                               float* __restrict__ h0,
                                               int ntile_rows) {
    __shared__ __align__(16) unsigned short xs[4][16 * CH];
    int lane = threadIdx.x & 31;
    int wave = threadIdx.x >> 5;
    int tile = blockIdx.x * 4 + wave;
    if (tile >= ntile_rows) return;
    unsigned short* my = xs[wave];

    // stage 16 rows of x as bf16 (2 lanes per row, 128 cols each)
    int row = lane >> 1;
    int c0 = (lane & 1) * 128;
    const float* xr = x + (size_t)(tile * 16 + row) * CH;
    for (int c = c0; c < c0 + 128; c += 4) {
        float4 v = *(const float4*)(xr + c);
        ushort4 u;
        u.x = f2bf(v.x); u.y = f2bf(v.y); u.z = f2bf(v.z); u.w = f2bf(v.w);
        *(ushort4*)&my[row * CH + c] = u;
    }

    int mrow  = lane & 15;
    int hi    = (lane >= 16) ? 8 : 0;
    int outM0 = (lane < 16) ? 0 : 8;
#pragma unroll 1
    for (int nt = 0; nt < 16; ++nt) {
        FragC acc = {};
#pragma unroll
        for (int kt = 0; kt < 8; ++kt) {
            FragAB a, b;
            const unsigned short* ap = &my[mrow * CH + kt * 32 + hi];
            a.q[0] = *(const uint4*)(ap);
            a.q[1] = *(const uint4*)(ap + 16);
            const unsigned short* bp = pW + (size_t)(kt * 16 + nt) * 512 + lane * 16;
            b.q[0] = *(const uint4*)(bp);
            b.q[1] = *(const uint4*)(bp + 8);
            acc.v = __builtin_amdgcn_wmma_f32_16x16x32_bf16(
                        false, a.v, false, b.v, (short)0, acc.v, false, false);
        }
        float* orow = h0 + (size_t)(tile * 16) * CH + nt * 16 + (lane & 15);
#pragma unroll
        for (int j = 0; j < 8; ++j)
            orow[(size_t)(outM0 + j) * CH] = acc.f[j];
    }
}

// ---------------------------------------------------------------------------
// h = b_gcn + dinv[i]^2 * h0[i]   (self-loop term; initializes h for atomics)
// ---------------------------------------------------------------------------
__global__ void h_init(const float* __restrict__ h0, const float* __restrict__ dinv,
                       const float* __restrict__ bgcn, float* __restrict__ h) {
    int idx = blockIdx.x * blockDim.x + threadIdx.x;   // N_NODES*64 float4 slots
    if (idx >= N_NODES * 64) return;
    int node = idx >> 6;
    int c = (idx & 63) << 2;
    float di = dinv[node];
    float nrm = di * di;
    float4 v = *(const float4*)(h0 + (size_t)node * CH + c);
    float4 b = *(const float4*)(bgcn + c);
    float4 o;
    o.x = b.x + v.x * nrm; o.y = b.y + v.y * nrm;
    o.z = b.z + v.z * nrm; o.w = b.w + v.w * nrm;
    *(float4*)(h + (size_t)node * CH + c) = o;
}

// ---------------------------------------------------------------------------
// h[dst] += norm * h0[src] over all edges (64 threads/edge, float4 each)
// ---------------------------------------------------------------------------
__global__ void aggregate(const float* __restrict__ h0, const float* __restrict__ dinv,
                          const int* __restrict__ ei, float* __restrict__ h) {
    int idx = blockIdx.x * blockDim.x + threadIdx.x;   // N_EDGES*64
    if (idx >= N_EDGES * 64) return;
    int e = idx >> 6;
    int c = (idx & 63) << 2;
    int s = ei[e];
    int d = ei[N_EDGES + e];
    float nrm = dinv[s] * dinv[d];
    float4 v = *(const float4*)(h0 + (size_t)s * CH + c);
    float* hd = h + (size_t)d * CH + c;
    atomicAdd(hd + 0, v.x * nrm);
    atomicAdd(hd + 1, v.y * nrm);
    atomicAdd(hd + 2, v.z * nrm);
    atomicAdd(hd + 3, v.w * nrm);
}

// ---------------------------------------------------------------------------
// Fused MLP head: d = h[dst]-h[src]; relu(d@W1+b1); relu(@W2+b2); @W3+b3
// One wave handles 16 query edges; layers 1-2 via bf16 WMMA, layer 3 scalar fp32.
// ---------------------------------------------------------------------------
__global__ __launch_bounds__(128) void mlp_kernel(const float* __restrict__ h,
        const int* __restrict__ eli,
        const unsigned short* __restrict__ pW1, const float* __restrict__ b1,
        const unsigned short* __restrict__ pW2, const float* __restrict__ b2,
        const float* __restrict__ W3, const float* __restrict__ b3,
        float* __restrict__ out) {
    __shared__ __align__(16) unsigned short dt[4][16 * CH];    // bf16 d tile
    __shared__ __align__(16) unsigned short a1[4][16 * HID1];  // bf16 act1
    __shared__ __align__(16) float          a2[4][16 * HID2];  // f32 act2
    int lane = threadIdx.x & 31;
    int wave = threadIdx.x >> 5;
    int tile = blockIdx.x * 4 + wave;          // 25000 tiles, grid*4 == 25000
    int q0 = tile * 16;
    unsigned short* mydt = dt[wave];
    unsigned short* mya1 = a1[wave];
    float*          mya2 = a2[wave];

    // stage d = h[dst] - h[src] as bf16
    int row = lane >> 1;
    int c0 = (lane & 1) * 128;
    int q = q0 + row;
    int s = eli[q];
    int d = eli[N_LABEL + q];
    const float* hs = h + (size_t)s * CH;
    const float* hd = h + (size_t)d * CH;
    for (int c = c0; c < c0 + 128; c += 4) {
        float4 a = *(const float4*)(hd + c);
        float4 b = *(const float4*)(hs + c);
        ushort4 u;
        u.x = f2bf(a.x - b.x); u.y = f2bf(a.y - b.y);
        u.z = f2bf(a.z - b.z); u.w = f2bf(a.w - b.w);
        *(ushort4*)&mydt[row * CH + c] = u;
    }

    int mrow  = lane & 15;
    int hi    = (lane >= 16) ? 8 : 0;
    int outM0 = (lane < 16) ? 0 : 8;
    int ncol  = lane & 15;

    // layer 1: 16x256 @ 256x128
#pragma unroll 1
    for (int nt = 0; nt < 8; ++nt) {
        FragC acc = {};
#pragma unroll
        for (int kt = 0; kt < 8; ++kt) {
            FragAB a, b;
            const unsigned short* ap = &mydt[mrow * CH + kt * 32 + hi];
            a.q[0] = *(const uint4*)(ap);
            a.q[1] = *(const uint4*)(ap + 16);
            const unsigned short* bp = pW1 + (size_t)(kt * 8 + nt) * 512 + lane * 16;
            b.q[0] = *(const uint4*)(bp);
            b.q[1] = *(const uint4*)(bp + 8);
            acc.v = __builtin_amdgcn_wmma_f32_16x16x32_bf16(
                        false, a.v, false, b.v, (short)0, acc.v, false, false);
        }
        float bias = b1[nt * 16 + ncol];
#pragma unroll
        for (int j = 0; j < 8; ++j) {
            float v = acc.f[j] + bias;
            v = v > 0.f ? v : 0.f;
            mya1[(outM0 + j) * HID1 + nt * 16 + ncol] = f2bf(v);
        }
    }

    // layer 2: 16x128 @ 128x32
#pragma unroll 1
    for (int nt = 0; nt < 2; ++nt) {
        FragC acc = {};
#pragma unroll
        for (int kt = 0; kt < 4; ++kt) {
            FragAB a, b;
            const unsigned short* ap = &mya1[mrow * HID1 + kt * 32 + hi];
            a.q[0] = *(const uint4*)(ap);
            a.q[1] = *(const uint4*)(ap + 16);
            const unsigned short* bp = pW2 + (size_t)(kt * 2 + nt) * 512 + lane * 16;
            b.q[0] = *(const uint4*)(bp);
            b.q[1] = *(const uint4*)(bp + 8);
            acc.v = __builtin_amdgcn_wmma_f32_16x16x32_bf16(
                        false, a.v, false, b.v, (short)0, acc.v, false, false);
        }
        float bias = b2[nt * 16 + ncol];
#pragma unroll
        for (int j = 0; j < 8; ++j) {
            float v = acc.f[j] + bias;
            v = v > 0.f ? v : 0.f;
            mya2[(outM0 + j) * HID2 + nt * 16 + ncol] = v;
        }
    }

    // layer 3: 16x32 @ 32x1 on lanes 0..15
    if (lane < 16) {
        float sum = b3[0];
#pragma unroll
        for (int cc = 0; cc < HID2; ++cc)
            sum += mya2[lane * HID2 + cc] * W3[cc];
        out[q0 + lane] = sum;
    }
}

// ---------------------------------------------------------------------------
extern "C" void kernel_launch(void* const* d_in, const int* in_sizes, int n_in,
                              void* d_out, int out_size, void* d_ws, size_t ws_size,
                              hipStream_t stream) {
    const float* x   = (const float*)d_in[0];
    const int*   ei  = (const int*)d_in[1];
    const int*   eli = (const int*)d_in[2];
    const float* Wg  = (const float*)d_in[3];
    const float* bg  = (const float*)d_in[4];
    const float* W1  = (const float*)d_in[5];
    const float* b1  = (const float*)d_in[6];
    const float* W2  = (const float*)d_in[7];
    const float* b2  = (const float*)d_in[8];
    const float* W3  = (const float*)d_in[9];
    const float* b3  = (const float*)d_in[10];
    float* out = (float*)d_out;

    char* ws = (char*)d_ws;
    float* h0 = (float*)ws;                     ws += (size_t)N_NODES * CH * 4;
    float* h  = (float*)ws;                     ws += (size_t)N_NODES * CH * 4;
    float* deg  = (float*)ws;                   ws += (size_t)N_NODES * 4;
    float* dinv = (float*)ws;                   ws += (size_t)N_NODES * 4;
    unsigned short* pWg = (unsigned short*)ws;  ws += (size_t)8 * 16 * 512 * 2;
    unsigned short* pW1 = (unsigned short*)ws;  ws += (size_t)8 * 8  * 512 * 2;
    unsigned short* pW2 = (unsigned short*)ws;  ws += (size_t)4 * 2  * 512 * 2;

    pack_weight<<<128, 32, 0, stream>>>(Wg, pWg, 256, 256);
    pack_weight<<<64,  32, 0, stream>>>(W1, pW1, 256, 128);
    pack_weight<<<8,   32, 0, stream>>>(W2, pW2, 128, 32);

    deg_init  <<<(N_NODES + 255) / 256, 256, 0, stream>>>(deg);
    deg_count <<<(N_EDGES + 255) / 256, 256, 0, stream>>>(ei, deg);
    dinv_kernel<<<(N_NODES + 255) / 256, 256, 0, stream>>>(deg, dinv);

    int row_tiles = N_NODES / 16;                         // 6250
    gemm_h0<<<(row_tiles + 3) / 4, 128, 0, stream>>>(x, pWg, h0, row_tiles);

    h_init   <<<(N_NODES * 64 + 255) / 256, 256, 0, stream>>>(h0, dinv, bg, h);
    aggregate<<<(N_EDGES * 64 + 255) / 256, 256, 0, stream>>>(h0, dinv, ei, h);

    mlp_kernel<<<N_LABEL / 64, 128, 0, stream>>>(h, eli, pW1, b1, pW2, b2, W3, b3, out);
}
